// EncoderRNN_7404523618352
// MI455X (gfx1250) — matrix operand order repaired and checked by reading, not compiled, not run on hardware
//
#include <hip/hip_runtime.h>
#include <hip/hip_bf16.h>

typedef __attribute__((ext_vector_type(16))) __bf16 bf16x16;
typedef __attribute__((ext_vector_type(8)))  float  f32x8;

#define S_LEN   1024
#define BATCH   32
#define HID     512
#define VOCAB   32000
#define NWG_DIR 32
#define THREADS 256

// workspace layout (bytes)
#define OFF_BH2  0ull
#define OFF_HBUF ((size_t)S_LEN * BATCH * HID * 4)                 // 67,108,864
#define OFF_BARS (OFF_HBUF + (size_t)2 * 2 * 2 * BATCH * HID * 2)  // + 262,144
#define OFF_EMBB (OFF_BARS + (size_t)2 * S_LEN * 2 * 4)            // + 16,384
// emb bf16: VOCAB*HID*2 = 32,768,000 -> total ws ~100.2 MB

// LDS layout (dynamic, 320KB total):
//   [0, 262144)      packed weights bf16, B-fragment order: [l][gate][kc][lane][16]
//   [262144, 327680) A buffer bf16 [32][1024]; first 8KB reused as f32 gate tile [32][64]

struct LstmParams {
  const float* Wih[2][2];
  const float* Whh[2][2];
  const float* bih[2][2];
  const float* bhh[2][2];
};

__device__ __forceinline__ float sigm(float x) { return 1.0f / (1.0f + __expf(-x)); }
__device__ __forceinline__ float tanh_fast(float x) { return 2.0f * sigm(2.0f * x) - 1.0f; }

__device__ __forceinline__ void grid_barrier(unsigned* ctr) {
  __threadfence();          // release all this-WG global stores to device scope
  __syncthreads();
  if (threadIdx.x == 0) {
    __hip_atomic_fetch_add(ctr, 1u, __ATOMIC_RELAXED, __HIP_MEMORY_SCOPE_AGENT);
    while (__hip_atomic_load(ctr, __ATOMIC_RELAXED, __HIP_MEMORY_SCOPE_AGENT) < (unsigned)NWG_DIR) {
      __builtin_amdgcn_s_sleep(1);
    }
  }
  __syncthreads();
  __threadfence();          // acquire: invalidate stale lines on both CUs of the WGP
}

// async copy of 128 contiguous bytes (8 x b128) from global bf16 to LDS
__device__ __forceinline__ void async_copy128(const __bf16* __restrict__ gsrc, __bf16* ldst) {
  unsigned ldso = (unsigned)(uintptr_t)(void*)ldst;   // low 32 bits = LDS byte offset
#pragma unroll
  for (int j = 0; j < 8; ++j) {
    unsigned long long ga = (unsigned long long)(uintptr_t)(gsrc + j * 8);
    asm volatile("global_load_async_to_lds_b128 %0, %1, off"
                 :: "v"(ldso + j * 16), "v"(ga) : "memory");
  }
}

__global__ void __launch_bounds__(THREADS, 1)
lstm_persistent(const int* __restrict__ widx, LstmParams p,
                float* __restrict__ out, void* __restrict__ ws) {
  extern __shared__ char smem[];
  __bf16* wlds = (__bf16*)smem;             // 256 KB resident weights (both layers)
  __bf16* Abuf = (__bf16*)(smem + 262144);  // 64 KB  [x | h_prev] staging
  float*  gbuf = (float*)(smem + 262144);   // 8 KB   gate-tile overlay on Abuf

  float*        bh2  = (float*)((char*)ws + OFF_BH2);
  __bf16*       hbuf = (__bf16*)((char*)ws + OFF_HBUF);   // [dir][layer][pp][B*H]
  unsigned*     bars = (unsigned*)((char*)ws + OFF_BARS); // [dir][s][2]
  const __bf16* embb = (const __bf16*)((char*)ws + OFF_EMBB);

  const int tid  = threadIdx.x;
  const int lane = tid & 31;
  const int wv   = tid >> 5;
  const int mt   = wv >> 2;   // batch-row tile: rows 0-15 / 16-31
  const int nt   = wv & 3;    // gate tile: i,f,g,o
  const int hi   = lane >> 4;
  const int ln   = lane & 15;

  const int dir = blockIdx.x >> 5;   // 0 = fwd, 1 = bwd
  const int wg  = blockIdx.x & 31;
  const int c0  = wg * 16;           // owned h-column block

  // ---- one-time: pack this WG's weight slice into LDS (bf16, B-fragment order) ----
  for (int l = 0; l < 2; ++l) {
    const float* Wih = p.Wih[dir][l];
    const float* Whh = p.Whh[dir][l];
    for (int idx = tid; idx < 4 * 32 * 32 * 16; idx += THREADS) {
      int e   =  idx        & 15;
      int lnn = (idx >> 4)  & 31;
      int kc  = (idx >> 9)  & 31;
      int gt  = (idx >> 14) & 3;
      int kk  = kc * 32 + (lnn >> 4) * 8 + (e & 7) + (e >> 3) * 16;   // B operand: lanes 0-15 K 0-7,16-23; 16-31 K 8-15,24-31
      int g   = gt * 512 + c0 + (lnn & 15);
      float v = (kk < 512) ? Wih[g * 512 + kk] : Whh[g * 512 + (kk - 512)];
      wlds[(size_t)l * 65536 + idx] = (__bf16)v;
    }
  }

  // per-thread combined biases for its h-column
  const int hc   = tid >> 4;     // 0..15
  const int hcol = c0 + hc;
  float bias[2][4];
  for (int l = 0; l < 2; ++l)
    for (int gt = 0; gt < 4; ++gt)
      bias[l][gt] = p.bih[dir][l][gt * 512 + hcol] + p.bhh[dir][l][gt * 512 + hcol];

  float c_state[2][2] = {{0.f, 0.f}, {0.f, 0.f}};   // [layer][batch pair]

  __syncthreads();

  const int row = tid >> 3;   // staging row 0..31
  const int seg = tid & 7;    // 64-column segment

#pragma unroll 1
  for (int s = 0; s < S_LEN; ++s) {
    const int t_in = dir ? (S_LEN - 1 - s) : s;
    const int pp   = s & 1;
#pragma unroll 1
    for (int l = 0; l < 2; ++l) {
      // ---- stage A = [x | h_prev] into LDS via async global->LDS DMA ----
      if (l == 0) {
        const int w = widx[t_in * BATCH + row];
        async_copy128(embb + (size_t)w * HID + seg * 64,
                      Abuf + row * 1024 + seg * 64);
      } else {
        const __bf16* h1new = hbuf + ((size_t)(dir * 2 + 0) * 2 + (pp ^ 1)) * BATCH * HID;
        async_copy128(h1new + row * HID + seg * 64,
                      Abuf + row * 1024 + seg * 64);
      }
      {
        const __bf16* hprev = hbuf + ((size_t)(dir * 2 + l) * 2 + pp) * BATCH * HID;
        async_copy128(hprev + row * HID + seg * 64,
                      Abuf + row * 1024 + 512 + seg * 64);
      }
      asm volatile("s_wait_asynccnt 0x0" ::: "memory");
      __syncthreads();

      // ---- WMMA: one 16x16 gate tile per wave, K = 1024 (two accumulators for ILP) ----
      f32x8 acc0 = {0.f, 0.f, 0.f, 0.f, 0.f, 0.f, 0.f, 0.f};
      f32x8 acc1 = {0.f, 0.f, 0.f, 0.f, 0.f, 0.f, 0.f, 0.f};
      const __bf16* arow = Abuf + (mt * 16 + ln) * 1024 + hi * 8;
      const __bf16* brow = wlds + (size_t)l * 65536 + (size_t)nt * 16384 + lane * 16;
#pragma unroll 8
      for (int kc = 0; kc < 32; kc += 2) {
        union { uint4 u[2]; bf16x16 v; } af0, bf0, af1, bf1;
        af0.u[0] = *(const uint4*)(arow + kc * 32);
        af0.u[1] = *(const uint4*)(arow + kc * 32 + 16);
        bf0.u[0] = *(const uint4*)(brow + kc * 512);
        bf0.u[1] = *(const uint4*)(brow + kc * 512 + 8);
        af1.u[0] = *(const uint4*)(arow + (kc + 1) * 32);
        af1.u[1] = *(const uint4*)(arow + (kc + 1) * 32 + 16);
        bf1.u[0] = *(const uint4*)(brow + (kc + 1) * 512);
        bf1.u[1] = *(const uint4*)(brow + (kc + 1) * 512 + 8);
        acc0 = __builtin_amdgcn_wmma_f32_16x16x32_bf16(false, af0.v, false, bf0.v,
                                                       (short)0, acc0, false, false);
        acc1 = __builtin_amdgcn_wmma_f32_16x16x32_bf16(false, af1.v, false, bf1.v,
                                                       (short)0, acc1, false, false);
      }
      f32x8 acc = acc0 + acc1;

      __syncthreads();   // all waves done reading A (rows 0-3 about to be overlaid)

      // pull h_prev + mask into registers before the overlay is written
      float hprev_r[2]; bool msk[2];
#pragma unroll
      for (int pq = 0; pq < 2; ++pq) {
        int b = (tid & 15) * 2 + pq;
        hprev_r[pq] = (float)Abuf[b * 1024 + 512 + hcol];
        msk[pq] = (widx[t_in * BATCH + b] == 1);
      }
      __syncthreads();

      // scatter D fragments into the f32 gate tile (D: lane = N, VGPR r -> M = r + 8*hi)
      {
        float* gdst  = gbuf + nt * 16 + ln;
        int    mbase = mt * 16 + hi * 8;
#pragma unroll
        for (int r = 0; r < 8; ++r) gdst[(mbase + r) * 64] = acc[r];
      }
      __syncthreads();

      // ---- elementwise LSTM for this thread's (b, hcol) pairs ----
      __bf16* hdst = hbuf + ((size_t)(dir * 2 + l) * 2 + (pp ^ 1)) * BATCH * HID;
#pragma unroll
      for (int pq = 0; pq < 2; ++pq) {
        int b = (tid & 15) * 2 + pq;
        float gi = gbuf[b * 64 + 0 * 16 + hc] + bias[l][0];
        float gf = gbuf[b * 64 + 1 * 16 + hc] + bias[l][1];
        float gg = gbuf[b * 64 + 2 * 16 + hc] + bias[l][2];
        float go = gbuf[b * 64 + 3 * 16 + hc] + bias[l][3];
        float cp = c_state[l][pq];
        float cn = sigm(gf) * cp + sigm(gi) * tanh_fast(gg);
        float hn = sigm(go) * tanh_fast(cn);
        if (msk[pq]) { cn = cp; hn = hprev_r[pq]; }
        c_state[l][pq] = cn;
        hdst[b * HID + hcol] = (__bf16)hn;
        if (l == 1) {
          float* od = (dir == 0) ? out : bh2;
          od[((size_t)t_in * BATCH + b) * HID + hcol] = hn;
        }
        if (dir == 1 && s == S_LEN - 1) {   // backward final carries (bh1,bc1,bh2,bc2)
          size_t base = (size_t)S_LEN * BATCH * HID + (size_t)l * 2 * BATCH * HID;
          out[base + (size_t)b * HID + hcol]               = hn;
          out[base + (size_t)BATCH * HID + b * HID + hcol] = cn;
        }
      }
      grid_barrier(&bars[(dir * S_LEN + s) * 2 + l]);
    }
  }
}

__global__ void init_ws(void* ws) {
  unsigned* hb = (unsigned*)((char*)ws + OFF_HBUF);   // hbuf + bars are contiguous
  size_t n = (262144 + 16384) / 4;
  for (size_t i = (size_t)blockIdx.x * blockDim.x + threadIdx.x; i < n;
       i += (size_t)gridDim.x * blockDim.x)
    hb[i] = 0u;
}

__global__ void emb_to_bf16(const float* __restrict__ emb, void* __restrict__ ws) {
  __bf16* dst = (__bf16*)((char*)ws + OFF_EMBB);
  size_t i = ((size_t)blockIdx.x * blockDim.x + threadIdx.x) * 4;   // VOCAB*HID total
  float4 f = *(const float4*)(emb + i);
  union { __bf16 b[4]; uint2 u; } t;
  t.b[0] = (__bf16)f.x; t.b[1] = (__bf16)f.y; t.b[2] = (__bf16)f.z; t.b[3] = (__bf16)f.w;
  *(uint2*)(dst + i) = t.u;
}

__global__ void combine_out(float* __restrict__ out, const float* __restrict__ bh2) {
  size_t i = (size_t)blockIdx.x * blockDim.x + threadIdx.x;   // float4 index
  float4*       o = (float4*)out;
  const float4* b = (const float4*)bh2;
  float4 ov = o[i], bv = b[i];
  ov.x += 0.5f * bv.x; ov.y += 0.5f * bv.y; ov.z += 0.5f * bv.z; ov.w += 0.5f * bv.w;
  o[i] = ov;
}

extern "C" void kernel_launch(void* const* d_in, const int* in_sizes, int n_in,
                              void* d_out, int out_size, void* d_ws, size_t ws_size,
                              hipStream_t stream) {
  (void)in_sizes; (void)n_in; (void)out_size; (void)ws_size;
  const int*   widx = (const int*)d_in[0];
  const float* emb  = (const float*)d_in[1];
  LstmParams p;
  const int base[2][2] = {{2, 6}, {10, 14}};   // fwd1, fwd2, bwd1, bwd2
  for (int d = 0; d < 2; ++d)
    for (int l = 0; l < 2; ++l) {
      p.Wih[d][l] = (const float*)d_in[base[d][l] + 0];
      p.Whh[d][l] = (const float*)d_in[base[d][l] + 1];
      p.bih[d][l] = (const float*)d_in[base[d][l] + 2];
      p.bhh[d][l] = (const float*)d_in[base[d][l] + 3];
    }
  float* out = (float*)d_out;

  init_ws<<<272, 256, 0, stream>>>(d_ws);
  emb_to_bf16<<<(VOCAB * HID) / (256 * 4), 256, 0, stream>>>(emb, d_ws);

  (void)hipFuncSetAttribute(reinterpret_cast<const void*>(&lstm_persistent),
                            hipFuncAttributeMaxDynamicSharedMemorySize, 320 * 1024);
  lstm_persistent<<<2 * NWG_DIR, THREADS, 320 * 1024, stream>>>(widx, p, out, d_ws);

  combine_out<<<16384, 256, 0, stream>>>(out, (const float*)d_ws);
}
